// RetentionLayer_76596446756875
// MI455X (gfx1250) — matrix operand order repaired and checked by reading, not compiled, not run on hardware
//
#include <hip/hip_runtime.h>

typedef __attribute__((ext_vector_type(16))) __bf16 v16bf;
typedef __attribute__((ext_vector_type(8)))  float  v8f;
typedef __attribute__((ext_vector_type(4)))  int    v4i_;

#define D_EMB  1024
#define NHEAD  16
#define DHEAD  64
#define SEQ    2048
#define NBATCH 2
#define NTOK   (NBATCH * SEQ)   // 4096

union FragBF { v16bf v; uint4 u[2]; };

static __device__ inline v8f zero8() {
  v8f z;
#pragma unroll
  for (int i = 0; i < 8; ++i) z[i] = 0.0f;
  return z;
}

// A fragment (16x32, MxK bf16): lane%16 = row M; low lanes hold K {0..7,16..23},
// high lanes K {8..15,24..31} (CDNA5 ISA 7.12.2, 16-bit A layout).
static __device__ inline v16bf load_a(const __bf16* rowbase, int lhi) {
  const __bf16* p = rowbase + (lhi ? 8 : 0);
  FragBF f;
  f.u[0] = *reinterpret_cast<const uint4*>(p);
  f.u[1] = *reinterpret_cast<const uint4*>(p + 16);
  return f.v;
}

// B fragment (32x16, KxN bf16) from K-contiguous memory: lane%16 = column N;
// low lanes K {0..7,16..23}, high lanes K {8..15,24..31}.
static __device__ inline v16bf load_b(const __bf16* colbase, int lhi) {
  const __bf16* p = colbase + (lhi ? 8 : 0);
  FragBF f;
  f.u[0] = *reinterpret_cast<const uint4*>(p);
  f.u[1] = *reinterpret_cast<const uint4*>(p + 16);
  return f.v;
}

static __device__ inline v8f wmma_bf16(v16bf a, v16bf b, v8f c) {
  return __builtin_amdgcn_wmma_f32_16x16x32_bf16(
      false, a, false, b, (short)0, c, false, false);
}

// --- CDNA5 async global->LDS copy (ASYNCcnt-tracked), 16B per lane ----------
typedef __attribute__((address_space(1))) v4i_* gv4i_p;  // global
typedef __attribute__((address_space(3))) v4i_* lv4i_p;  // LDS

static __device__ inline void async_copy_b128(const __bf16* g, __bf16* l) {
#if __has_builtin(__builtin_amdgcn_global_load_async_to_lds_b128)
  __builtin_amdgcn_global_load_async_to_lds_b128((gv4i_p)g, (lv4i_p)l, 0, 0);
#else
  unsigned loff = (unsigned)(unsigned long long)l;  // LDS aperture: addr[31:0]
  asm volatile("global_load_async_to_lds_b128 %0, %1, off"
               :
               : "v"(loff), "v"(g)
               : "memory");
#endif
}

static __device__ inline void wait_async0() {
#if __has_builtin(__builtin_amdgcn_s_wait_asynccnt)
  __builtin_amdgcn_s_wait_asynccnt(0);
#else
  asm volatile("s_wait_asynccnt 0x0" ::: "memory");
#endif
}

// ---------------------------------------------------------------------------
// fp32 -> bf16 conversion (4 elems/thread)
// ---------------------------------------------------------------------------
__global__ void cvt_f32_bf16(const float* __restrict__ in,
                             __bf16* __restrict__ out, int n) {
  int i = (blockIdx.x * blockDim.x + threadIdx.x) * 4;
  if (i + 3 < n) {
    float4 f = *reinterpret_cast<const float4*>(in + i);
    __bf16 t[4] = {(__bf16)f.x, (__bf16)f.y, (__bf16)f.z, (__bf16)f.w};
    *reinterpret_cast<uint2*>(out + i) = *reinterpret_cast<uint2*>(t);
  }
}

// ---------------------------------------------------------------------------
// GEMM: C[M,N] = A[M,K] * B[N,K]^T   (bf16 in, fp32 accumulate)
// 256 threads = 8 waves, block tile 128x128, wave tile 32x64 (2x4 WMMA accs).
// K staged through LDS in 32-wide slabs with double-buffered async copies:
// stage s+1's ASYNCcnt copies overlap stage s's 8 WMMAs.
// MODE 0: store bf16           (Q, K before RoPE)
// MODE 1: SiLU -> bf16         (gate)
// MODE 2: store fp32           (final output)
// MODE 3: store bf16 transposed to (b, h, d, s) layout   (V^T for retention)
// ---------------------------------------------------------------------------
template <int MODE>
__global__ __launch_bounds__(256) void gemm_bf16(
    const __bf16* __restrict__ A, const __bf16* __restrict__ Bm,
    void* __restrict__ Cout, int M, int N, int K) {
  __shared__ __align__(16) __bf16 Abuf[2][128 * 32];  // 8KB x2
  __shared__ __align__(16) __bf16 Bbuf[2][128 * 32];  // 8KB x2

  const int lane   = threadIdx.x & 31;
  const int lane16 = lane & 15;
  const int lhi    = lane >> 4;
  const int wave   = threadIdx.x >> 5;
  const int waveM  = wave >> 1;   // 0..3
  const int waveN  = wave & 1;    // 0..1
  const int mBlk = blockIdx.x * 128;
  const int nBlk = blockIdx.y * 128;

  v8f acc[2][4];
#pragma unroll
  for (int mt = 0; mt < 2; ++mt)
#pragma unroll
    for (int nt = 0; nt < 4; ++nt) acc[mt][nt] = zero8();

  // cooperative async stage of one 128x32 A slab + 128x32 B slab
  auto stage = [&](int buf, int k0) {
#pragma unroll
    for (int i = 0; i < 2; ++i) {
      int c   = (int)threadIdx.x + i * 256;  // 512 16B-chunks per matrix
      int row = c >> 2;                      // 4 chunks per 32-elem row
      int col = (c & 3) * 8;
      async_copy_b128(A + (size_t)(mBlk + row) * K + k0 + col,
                      &Abuf[buf][row * 32 + col]);
      async_copy_b128(Bm + (size_t)(nBlk + row) * K + k0 + col,
                      &Bbuf[buf][row * 32 + col]);
    }
  };

  const int nStages = K >> 5;
  stage(0, 0);
  for (int s = 0; s < nStages; ++s) {
    const int cur = s & 1;
    wait_async0();     // own stage-s copies complete
    __syncthreads();   // -> everyone's stage-s copies visible
    if (s + 1 < nStages) stage(cur ^ 1, (s + 1) << 5);  // overlap with compute

    const __bf16* aBase = &Abuf[cur][(waveM * 32 + lane16) * 32];
    v16bf a0 = load_a(aBase, lhi);
    v16bf a1 = load_a(aBase + 16 * 32, lhi);
#pragma unroll
    for (int nt = 0; nt < 4; ++nt) {
      v16bf b = load_b(&Bbuf[cur][(waveN * 64 + nt * 16 + lane16) * 32], lhi);
      acc[0][nt] = wmma_bf16(a0, b, acc[0][nt]);
      acc[1][nt] = wmma_bf16(a1, b, acc[1][nt]);
    }
    __syncthreads();   // all reads of `cur` done before stage s+2 overwrites it
  }

#pragma unroll
  for (int mt = 0; mt < 2; ++mt)
#pragma unroll
    for (int nt = 0; nt < 4; ++nt)
#pragma unroll
      for (int r = 0; r < 8; ++r) {
        int m = mBlk + waveM * 32 + mt * 16 + r + lhi * 8;
        int n = nBlk + waveN * 64 + nt * 16 + lane16;
        float v = acc[mt][nt][r];
        if (MODE == 0) {
          ((__bf16*)Cout)[(size_t)m * N + n] = (__bf16)v;
        } else if (MODE == 1) {
          v = v / (1.0f + __expf(-v));  // SiLU
          ((__bf16*)Cout)[(size_t)m * N + n] = (__bf16)v;
        } else if (MODE == 2) {
          ((float*)Cout)[(size_t)m * N + n] = v;
        } else {  // MODE 3: V^T : (b, h, d, s)
          int b = m >> 11, ss = m & (SEQ - 1);
          int h = n >> 6, d = n & (DHEAD - 1);
          ((__bf16*)Cout)[(((size_t)(b * NHEAD + h)) * DHEAD + d) * SEQ + ss] =
              (__bf16)v;
        }
      }
}

// ---------------------------------------------------------------------------
// RoPE in-place on bf16 Q and K. One thread per (token, head, j<32) pair.
// ---------------------------------------------------------------------------
__global__ void rope_kernel(__bf16* __restrict__ q, __bf16* __restrict__ k) {
  int tid = blockIdx.x * blockDim.x + threadIdx.x;
  if (tid >= NTOK * NHEAD * 32) return;
  int j   = tid & 31;
  int h   = (tid >> 5) & (NHEAD - 1);
  int tok = tid >> 9;
  int s   = tok & (SEQ - 1);
  // inv_freq = 10000^(-2j/64) = exp2(-j * log2(10000)/32)
  float inv = __builtin_amdgcn_exp2f(-(float)j * 0.41524101186092030f);
  float ang = (float)s * inv;
  float c = cosf(ang), si = sinf(ang);
  size_t i1 = (size_t)tok * D_EMB + h * DHEAD + j;
  size_t i2 = i1 + 32;
  float q1 = (float)q[i1], q2 = (float)q[i2];
  q[i1] = (__bf16)(q1 * c - q2 * si);
  q[i2] = (__bf16)(q2 * c + q1 * si);
  float k1 = (float)k[i1], k2 = (float)k[i2];
  k[i1] = (__bf16)(k1 * c - k2 * si);
  k[i2] = (__bf16)(k2 * c + k1 * si);
}

// ---------------------------------------------------------------------------
// Retention + GroupNorm + gate, fused. Grid (S/64, B*H), block = 4 waves.
// Each wave owns one 16-row s-tile, accumulating O(16 x 64) in 4 WMMA accs.
// ---------------------------------------------------------------------------
__global__ __launch_bounds__(128) void retention_kernel(
    const __bf16* __restrict__ q, const __bf16* __restrict__ k,
    const __bf16* __restrict__ vT, const __bf16* __restrict__ gate,
    const float* __restrict__ gnw, const float* __restrict__ gnb,
    __bf16* __restrict__ act) {
  __shared__ __align__(16) __bf16 Pld[4][16 * 32];  // per-wave P tile (bf16)

  const int lane   = threadIdx.x & 31;
  const int lane16 = lane & 15;
  const int lhi    = lane >> 4;
  const int wave   = threadIdx.x >> 5;
  const int bh = blockIdx.y;
  const int b  = bh >> 4;
  const int h  = bh & (NHEAD - 1);
  const int s0 = blockIdx.x * 64;
  const int sw = s0 + wave * 16;

  const float gamma = 1.0f - __builtin_amdgcn_exp2f(-5.0f - (float)h);
  const float l2g   = __builtin_amdgcn_logf(gamma) * 1.4426950408889634f;

  // Q A-fragments for this wave's 16 rows, both dh-chunks (loop invariant)
  const __bf16* qrow = q + (size_t)(b * SEQ + sw + lane16) * D_EMB + h * DHEAD;
  v16bf aq0 = load_a(qrow, lhi);
  v16bf aq1 = load_a(qrow + 32, lhi);

  v8f acc[4];
#pragma unroll
  for (int dt = 0; dt < 4; ++dt) acc[dt] = zero8();

  const int tEnd = s0 + 64;  // causal: decay mask zeros everything past s
  for (int t0 = 0; t0 < tEnd; t0 += 32) {
    // ---- P(16x32) = Q * K^T for this wave's s-tile -----------------------
    v8f p0 = zero8(), p1 = zero8();
    const __bf16* kr0 =
        k + (size_t)(b * SEQ + t0 + lane16) * D_EMB + h * DHEAD;
    const __bf16* kr1 = kr0 + (size_t)16 * D_EMB;
    p0 = wmma_bf16(aq0, load_b(kr0, lhi), p0);
    p0 = wmma_bf16(aq1, load_b(kr0 + 32, lhi), p0);
    p1 = wmma_bf16(aq0, load_b(kr1, lhi), p1);
    p1 = wmma_bf16(aq1, load_b(kr1 + 32, lhi), p1);

    // ---- scale + causal decay (branchless), re-stripe via LDS ------------
#pragma unroll
    for (int r = 0; r < 8; ++r) {
      int sa = sw + lhi * 8 + r;
      int ta = t0 + lane16;
      int d0 = sa - ta, d1 = d0 - 16;
      float e0 = __builtin_amdgcn_exp2f((float)d0 * l2g) * 0.125f;
      float e1 = __builtin_amdgcn_exp2f((float)d1 * l2g) * 0.125f;
      float w0 = (d0 >= 0) ? e0 : 0.0f;  // v_cndmask, no branch
      float w1 = (d1 >= 0) ? e1 : 0.0f;
      Pld[wave][(r + 8 * lhi) * 32 + lane16]      = (__bf16)(p0[r] * w0);
      Pld[wave][(r + 8 * lhi) * 32 + 16 + lane16] = (__bf16)(p1[r] * w1);
    }
    __syncthreads();  // store -> cross-lane load ordering

    // ---- O += P * V  (A from LDS, B from pre-transposed V) ---------------
    // (no trailing barrier: Pld region is wave-private and same-wave LDS ops
    //  are processed in order, so next iteration's stores cannot pass this
    //  iteration's loads)
    v16bf ap = load_a(&Pld[wave][lane16 * 32], lhi);
#pragma unroll
    for (int dt = 0; dt < 4; ++dt) {
      const __bf16* vr =
          vT + ((size_t)bh * DHEAD + dt * 16 + lane16) * SEQ + t0;
      acc[dt] = wmma_bf16(ap, load_b(vr, lhi), acc[dt]);
    }
  }

  // ---- fused GroupNorm (over Dh=64) + affine + gate ----------------------
  float wv[4], bv[4];
#pragma unroll
  for (int dt = 0; dt < 4; ++dt) {
    int n = h * DHEAD + dt * 16 + lane16;
    wv[dt] = gnw[n];
    bv[dt] = gnb[n];
  }
#pragma unroll
  for (int r = 0; r < 8; ++r) {
    float sum = 0.0f, sq = 0.0f;
#pragma unroll
    for (int dt = 0; dt < 4; ++dt) {
      float x = acc[dt][r];
      sum += x;
      sq += x * x;
    }
    // reduce across the 16-lane half holding this row's 16 N-columns
#pragma unroll
    for (int off = 8; off; off >>= 1) {
      sum += __shfl_xor(sum, off, 32);
      sq  += __shfl_xor(sq, off, 32);
    }
    float mean = sum * (1.0f / 64.0f);
    float var  = sq * (1.0f / 64.0f) - mean * mean;
    float rstd = rsqrtf(var + 1e-5f);
    int tok = b * SEQ + sw + r + lhi * 8;
#pragma unroll
    for (int dt = 0; dt < 4; ++dt) {
      int n = h * DHEAD + dt * 16 + lane16;
      float g = (float)gate[(size_t)tok * D_EMB + n];
      float val = ((acc[dt][r] - mean) * rstd * wv[dt] + bv[dt]) * g;
      act[(size_t)tok * D_EMB + n] = (__bf16)val;
    }
  }
}

// ---------------------------------------------------------------------------
extern "C" void kernel_launch(void* const* d_in, const int* in_sizes, int n_in,
                              void* d_out, int out_size, void* d_ws,
                              size_t ws_size, hipStream_t stream) {
  (void)in_sizes; (void)n_in; (void)out_size; (void)ws_size;
  const float* x   = (const float*)d_in[0];
  const float* Wq  = (const float*)d_in[1];
  const float* Wk  = (const float*)d_in[2];
  const float* Wv  = (const float*)d_in[3];
  const float* Wg  = (const float*)d_in[4];
  const float* Wo  = (const float*)d_in[5];
  const float* gnw = (const float*)d_in[6];
  const float* gnb = (const float*)d_in[7];

  const size_t TOKD = (size_t)NTOK * D_EMB;     // 4M elems
  const size_t WSZ  = (size_t)D_EMB * D_EMB;    // 1M elems
  char* p = (char*)d_ws;
  __bf16* xb   = (__bf16*)p; p += TOKD * 2;
  __bf16* wqb  = (__bf16*)p; p += WSZ * 2;
  __bf16* wkb  = (__bf16*)p; p += WSZ * 2;
  __bf16* wvb  = (__bf16*)p; p += WSZ * 2;
  __bf16* wgb  = (__bf16*)p; p += WSZ * 2;
  __bf16* wob  = (__bf16*)p; p += WSZ * 2;
  __bf16* qb   = (__bf16*)p; p += TOKD * 2;
  __bf16* kb   = (__bf16*)p; p += TOKD * 2;
  __bf16* vTb  = (__bf16*)p; p += TOKD * 2;
  __bf16* gtb  = (__bf16*)p; p += TOKD * 2;
  __bf16* actb = (__bf16*)p; p += TOKD * 2;

  // bf16 conversions
  cvt_f32_bf16<<<(int)(TOKD / 4 + 255) / 256, 256, 0, stream>>>(x, xb, (int)TOKD);
  cvt_f32_bf16<<<(int)(WSZ / 4 + 255) / 256, 256, 0, stream>>>(Wq, wqb, (int)WSZ);
  cvt_f32_bf16<<<(int)(WSZ / 4 + 255) / 256, 256, 0, stream>>>(Wk, wkb, (int)WSZ);
  cvt_f32_bf16<<<(int)(WSZ / 4 + 255) / 256, 256, 0, stream>>>(Wv, wvb, (int)WSZ);
  cvt_f32_bf16<<<(int)(WSZ / 4 + 255) / 256, 256, 0, stream>>>(Wg, wgb, (int)WSZ);
  cvt_f32_bf16<<<(int)(WSZ / 4 + 255) / 256, 256, 0, stream>>>(Wo, wob, (int)WSZ);

  // projections
  dim3 gg(NTOK / 128, D_EMB / 128);
  gemm_bf16<0><<<gg, 256, 0, stream>>>(xb, wqb, qb, NTOK, D_EMB, D_EMB);
  gemm_bf16<0><<<gg, 256, 0, stream>>>(xb, wkb, kb, NTOK, D_EMB, D_EMB);
  gemm_bf16<3><<<gg, 256, 0, stream>>>(xb, wvb, vTb, NTOK, D_EMB, D_EMB);
  gemm_bf16<1><<<gg, 256, 0, stream>>>(xb, wgb, gtb, NTOK, D_EMB, D_EMB);

  // RoPE on Q, K
  int nR = NTOK * NHEAD * 32;
  rope_kernel<<<(nR + 255) / 256, 256, 0, stream>>>(qb, kb);

  // fused retention + groupnorm + gate
  dim3 ga(SEQ / 64, NBATCH * NHEAD);
  retention_kernel<<<ga, 128, 0, stream>>>(qb, kb, vTb, gtb, gnw, gnb, actb);

  // output projection (fp32 result)
  gemm_bf16<2><<<gg, 256, 0, stream>>>(actb, wob, (float*)d_out, NTOK, D_EMB,
                                       D_EMB);
}
